// LSTM_13829794693247
// MI455X (gfx1250) — matrix operand order repaired and checked by reading, not compiled, not run on hardware
//
#include <hip/hip_runtime.h>
#include <hip/hip_bf16.h>
#include <math.h>

// ---------------- problem constants (from reference) ----------------
#define T_STEPS 512
#define BATCH   64
#define IN_DIM  512
#define HID     1024
#define KTOT    (IN_DIM + HID)   // 1536 : concatenated [x | h] K dimension
#define NG      (4 * HID)        // 4096 : 4 gates stacked along N
#define APAD    8                // bf16 pad per LDS A row (bank-conflict free)
#define AROW    (KTOT + APAD)    // 1544 elements per LDS A row

typedef __attribute__((ext_vector_type(16))) __bf16 v16bf;
typedef __attribute__((ext_vector_type(8)))  __bf16 v8bf;
typedef __attribute__((ext_vector_type(8)))  float  v8f;

// ---------------- prep kernels ----------------

// fp32 -> bf16 convert of x, 4 elements per thread
__global__ void cvt_x_kernel(const float* __restrict__ x, __bf16* __restrict__ xb, int n) {
    int i = (blockIdx.x * blockDim.x + threadIdx.x) * 4;
    if (i + 3 < n) {
        float4 v = *(const float4*)(x + i);
        xb[i + 0] = (__bf16)v.x;
        xb[i + 1] = (__bf16)v.y;
        xb[i + 2] = (__bf16)v.z;
        xb[i + 3] = (__bf16)v.w;
    }
}

// Pack 8 weight matrices into W_cat[NG, KTOT] (bf16, row-major).
// Row n = g*HID + hh ; col k: k<IN_DIM -> w_i{g}[hh,k], else w_h{g}[hh,k-IN_DIM]
__global__ void pack_w_kernel(const float* __restrict__ wii, const float* __restrict__ wif,
                              const float* __restrict__ wig, const float* __restrict__ wio,
                              const float* __restrict__ whi, const float* __restrict__ whf,
                              const float* __restrict__ whg, const float* __restrict__ who,
                              __bf16* __restrict__ wcat) {
    int idx = blockIdx.x * blockDim.x + threadIdx.x;        // over NG*KTOT
    if (idx >= NG * KTOT) return;
    int n  = idx / KTOT;
    int k  = idx - n * KTOT;
    int g  = n >> 10;          // gate
    int hh = n & (HID - 1);    // hidden row
    float v;
    if (k < IN_DIM) {
        const float* w = (g == 0) ? wii : (g == 1) ? wif : (g == 2) ? wig : wio;
        v = w[hh * IN_DIM + k];
    } else {
        const float* w = (g == 0) ? whi : (g == 1) ? whf : (g == 2) ? whg : who;
        v = w[hh * HID + (k - IN_DIM)];
    }
    wcat[idx] = (__bf16)v;
}

// bias_cat[n] = b_i{g}[hh] + b_h{g}[hh]
__global__ void pack_bias_kernel(const float* __restrict__ bii, const float* __restrict__ bif,
                                 const float* __restrict__ big, const float* __restrict__ bio,
                                 const float* __restrict__ bhi, const float* __restrict__ bhf,
                                 const float* __restrict__ bhg, const float* __restrict__ bho,
                                 float* __restrict__ bias) {
    int n = blockIdx.x * blockDim.x + threadIdx.x;
    if (n >= NG) return;
    int g  = n >> 10;
    int hh = n & (HID - 1);
    const float* bi = (g == 0) ? bii : (g == 1) ? bif : (g == 2) ? big : bio;
    const float* bh = (g == 0) ? bhi : (g == 1) ? bhf : (g == 2) ? bhg : bho;
    bias[n] = bi[hh] + bh[hh];
}

// zero h0 (bf16) and c (f32)
__global__ void init_state_kernel(__bf16* __restrict__ h0, float* __restrict__ c) {
    int i = blockIdx.x * blockDim.x + threadIdx.x;
    if (i < BATCH * HID) {
        h0[i] = (__bf16)0.0f;
        c[i]  = 0.0f;
    }
}

__device__ __forceinline__ float sigmoidf_fast(float x) {
    return 1.0f / (1.0f + __expf(-x));
}

// ---------------- fused recurrent step ----------------
// grid: 256 blocks = 4 batch-tiles x 64 hidden-blocks ; block: 128 threads = 4 waves
// Stage the workgroup's A panel (16 rows x K=1536, bf16 = 48KB) into LDS once,
// then wave g computes gate g's 16x16 tile with A from LDS and B streamed from
// the L2-resident packed weight matrix.
__global__ __launch_bounds__(128) void lstm_step_kernel(
        const __bf16* __restrict__ xb,     // [T,B,IN_DIM] bf16
        const __bf16* __restrict__ wcat,   // [NG, KTOT]   bf16
        const float*  __restrict__ bias,   // [NG]
        const __bf16* __restrict__ h_in,   // [B,HID] bf16 (ping)
        __bf16*       __restrict__ h_out,  // [B,HID] bf16 (pong)
        float*        __restrict__ c_state,// [B,HID] f32, in-place
        float*        __restrict__ out,    // d_out: [T,B,HID] ++ h_T ++ c_T
        int t) {
    __shared__ __bf16 a_lds[16 * AROW];    // 49,408 B : A panel [16][1536(+8)]
    __shared__ float  g_lds[4][16][17];    // gate tiles

    const int lane   = threadIdx.x & 31;
    const int wave   = threadIdx.x >> 5;        // == gate index
    const int m_tile = blockIdx.x & 3;          // batch tile (16 rows)
    const int hblk   = blockIdx.x >> 2;         // hidden-col block (16 cols)

    // ---- cooperative staging of A = [x_t | h_{t-1}] rows into LDS ----
    {
        const __bf16* xrow_base = xb + ((size_t)t * BATCH + m_tile * 16) * IN_DIM;
        const __bf16* hrow_base = h_in + (size_t)(m_tile * 16) * HID;
        // 16*1536 bf16 in 16-byte chunks: 3072 chunks / 128 threads = 24 iters
        for (int idx = threadIdx.x * 8; idx < 16 * KTOT; idx += 128 * 8) {
            const int r = idx / KTOT;
            const int k = idx - r * KTOT;           // multiple of 8; 512 | chunking
            v8bf v;
            if (k < IN_DIM) v = *(const v8bf*)(xrow_base + r * IN_DIM + k);
            else            v = *(const v8bf*)(hrow_base + r * HID + (k - IN_DIM));
            *(v8bf*)(&a_lds[r * AROW + k]) = v;
        }
    }
    __syncthreads();

    // A-fragment addressing: lane L holds row m = L&15,
    // 16 contiguous k at (L>>4)*16 within each K=32 step (matches B)
    const int mrow  = lane & 15;
    const int klane = (lane >> 4) * 16;
    const __bf16* arow = &a_lds[mrow * AROW];

    // B-fragment addressing: lane L holds col n = L&15,
    // 16 contiguous k at (L>>4)*16 within each K=32 step
    const int ncol  = hblk * 16 + (lane & 15);
    const __bf16* wrow = wcat + (size_t)(wave * HID + ncol) * KTOT;
    const int     kb_b = (lane >> 4) * 16;

    v8f acc = {};
#pragma unroll 4
    for (int kb = 0; kb < KTOT; kb += 32) {
        v8bf a_lo = *(const v8bf*)(arow + kb + klane);          // ds_load_b128
        v8bf a_hi = *(const v8bf*)(arow + kb + klane + 8);      // ds_load_b128
        v16bf a;
#pragma unroll
        for (int j = 0; j < 8; ++j) { a[j] = a_lo[j]; a[8 + j] = a_hi[j]; }
        v16bf bm = *(const v16bf*)(wrow + kb + kb_b);           // 2x global_load_b128
        acc = __builtin_amdgcn_wmma_f32_16x16x32_bf16(
                  false, a, false, bm, (short)0, acc, false, false);
    }

    // C/D layout: VGPR r -> row (L>>4)*8 + r, col L&15
    const int crow0 = (lane >> 4) * 8;
#pragma unroll
    for (int r = 0; r < 8; ++r)
        g_lds[wave][crow0 + r][lane & 15] = acc[r];
    __syncthreads();

    // pointwise gate math: 128 threads cover 16x16 = 256 elements
    for (int e = threadIdx.x; e < 256; e += 128) {
        const int m  = e >> 4;
        const int hh = e & 15;
        const int bb = m_tile * 16 + m;
        const int h  = hblk * 16 + hh;

        float zi = g_lds[0][m][hh] + bias[h];
        float zf = g_lds[1][m][hh] + bias[HID + h];
        float zg = g_lds[2][m][hh] + bias[2 * HID + h];
        float zo = g_lds[3][m][hh] + bias[3 * HID + h];

        float ig = sigmoidf_fast(zi);
        float fg = sigmoidf_fast(zf);
        float gg = tanhf(zg);
        float og = sigmoidf_fast(zo);

        const size_t idx = (size_t)bb * HID + h;
        float c_new = fg * c_state[idx] + ig * gg;
        float h_new = og * tanhf(c_new);

        c_state[idx] = c_new;
        h_out[idx]   = (__bf16)h_new;
        out[(size_t)t * BATCH * HID + idx] = h_new;
        if (t == T_STEPS - 1) {
            out[(size_t)T_STEPS * BATCH * HID + idx] = h_new;                       // h_T
            out[(size_t)T_STEPS * BATCH * HID + (size_t)BATCH * HID + idx] = c_new; // c_T
        }
    }
}

// ---------------- host side ----------------
extern "C" void kernel_launch(void* const* d_in, const int* in_sizes, int n_in,
                              void* d_out, int out_size, void* d_ws, size_t ws_size,
                              hipStream_t stream) {
    // input order: x, (w_ii,b_ii),(w_if,b_if),(w_ig,b_ig),(w_io,b_io),
    //              (w_hi,b_hi),(w_hf,b_hf),(w_hg,b_hg),(w_ho,b_ho)
    const float* x    = (const float*)d_in[0];
    const float* w_ii = (const float*)d_in[1];
    const float* b_ii = (const float*)d_in[2];
    const float* w_if = (const float*)d_in[3];
    const float* b_if = (const float*)d_in[4];
    const float* w_ig = (const float*)d_in[5];
    const float* b_ig = (const float*)d_in[6];
    const float* w_io = (const float*)d_in[7];
    const float* b_io = (const float*)d_in[8];
    const float* w_hi = (const float*)d_in[9];
    const float* b_hi = (const float*)d_in[10];
    const float* w_hf = (const float*)d_in[11];
    const float* b_hf = (const float*)d_in[12];
    const float* w_hg = (const float*)d_in[13];
    const float* b_hg = (const float*)d_in[14];
    const float* w_ho = (const float*)d_in[15];
    const float* b_ho = (const float*)d_in[16];

    float* out = (float*)d_out;

    // workspace carve-out (256B aligned slices)
    char*  ws  = (char*)d_ws;
    size_t off = 0;
    auto carve = [&](size_t bytes) -> void* {
        void* p = ws + off;
        off += (bytes + 255) & ~(size_t)255;
        return p;
    };
    __bf16* xb   = (__bf16*)carve((size_t)T_STEPS * BATCH * IN_DIM * sizeof(__bf16));
    __bf16* wcat = (__bf16*)carve((size_t)NG * KTOT * sizeof(__bf16));
    float*  bias = (float*) carve((size_t)NG * sizeof(float));
    __bf16* h0   = (__bf16*)carve((size_t)BATCH * HID * sizeof(__bf16));
    __bf16* h1   = (__bf16*)carve((size_t)BATCH * HID * sizeof(__bf16));
    float*  c    = (float*) carve((size_t)BATCH * HID * sizeof(float));
    (void)ws_size; (void)in_sizes; (void)n_in; (void)out_size;

    // prep
    {
        int n = T_STEPS * BATCH * IN_DIM;                 // divisible by 4
        cvt_x_kernel<<<(n / 4 + 255) / 256, 256, 0, stream>>>(x, xb, n);
    }
    {
        int n = NG * KTOT;
        pack_w_kernel<<<(n + 255) / 256, 256, 0, stream>>>(
            w_ii, w_if, w_ig, w_io, w_hi, w_hf, w_hg, w_ho, wcat);
    }
    pack_bias_kernel<<<(NG + 255) / 256, 256, 0, stream>>>(
        b_ii, b_if, b_ig, b_io, b_hi, b_hf, b_hg, b_ho, bias);
    init_state_kernel<<<(BATCH * HID + 255) / 256, 256, 0, stream>>>(h0, c);

    // sequential scan: one fused GEMM+gates kernel per timestep (stream-ordered)
    for (int t = 0; t < T_STEPS; ++t) {
        const __bf16* h_in  = (t & 1) ? h1 : h0;
        __bf16*       h_out = (t & 1) ? h0 : h1;
        lstm_step_kernel<<<256, 128, 0, stream>>>(xb, wcat, bias, h_in, h_out, c, out, t);
    }
}
